// RNN_7112465842328
// MI455X (gfx1250) — matrix operand order repaired and checked by reading, not compiled, not run on hardware
//
#include <hip/hip_runtime.h>
#include <hip/hip_bf16.h>
#include <math.h>

#define H 1024
#define B 128
#define T 256
#define L 2

typedef __attribute__((ext_vector_type(16))) __bf16 v16bf;
typedef __attribute__((ext_vector_type(8)))  __bf16 v8bf;
typedef __attribute__((ext_vector_type(8)))  float  v8f;

union BFV { v16bf v; v8bf h[2]; };

// Accumulate a 16x32 f32 tile (two 16x16 accumulators) over K = H with bf16
// WMMA, reusing each A tile across both B column-tiles (3 loads / WMMA).
// Arow: this lane's A row (row-major, ld = H).
// Bcol0: this lane's B column for the first 16-wide N tile (W^T row-major).
__device__ __forceinline__ void wmma_bb2(const __bf16* __restrict__ Arow,
                                         const __bf16* __restrict__ Bcol0,
                                         v8f* __restrict__ acc0,
                                         v8f* __restrict__ acc1, int half) {
  const __bf16* Bcol1 = Bcol0 + (size_t)16 * H;
#pragma unroll 4
  for (int k0 = 0; k0 < H; k0 += 32) {
    BFV a, b0, b1;
    // A 16x32 bf16: lanes 0-15 -> K0..15; lanes 16-31 -> K16..31
    a.h[0] = *(const v8bf*)(Arow + k0 + half * 16);
    a.h[1] = *(const v8bf*)(Arow + k0 + half * 16 + 8);
    // B 32x16 bf16: lanes 0-15 -> K0..15; lanes 16-31 -> K16..31
    b0.h[0] = *(const v8bf*)(Bcol0 + k0 + half * 16);
    b0.h[1] = *(const v8bf*)(Bcol0 + k0 + half * 16 + 8);
    b1.h[0] = *(const v8bf*)(Bcol1 + k0 + half * 16);
    b1.h[1] = *(const v8bf*)(Bcol1 + k0 + half * 16 + 8);
    *acc0 = __builtin_amdgcn_wmma_f32_16x16x32_bf16(false, a.v, false, b0.v,
                                                    (short)0, *acc0, false, false);
    *acc1 = __builtin_amdgcn_wmma_f32_16x16x32_bf16(false, a.v, false, b1.v,
                                                    (short)0, *acc1, false, false);
  }
}

// Transpose+convert the 4 H x H weight matrices to bf16 (one-time).
__global__ __launch_bounds__(256) void convert_weights_kernel(
    const float* __restrict__ Wi, const float* __restrict__ Wh,
    __bf16* __restrict__ Wi0T, __bf16* __restrict__ Wh0T,
    __bf16* __restrict__ Wi1T, __bf16* __restrict__ Wh1T) {
  size_t idx = (size_t)blockIdx.x * blockDim.x + threadIdx.x;  // 4 * H * H
  int mat = (int)(idx >> 20);
  int o = (int)(idx & ((1u << 20) - 1));
  int n = o >> 10, k = o & (H - 1);
  const float* src;
  __bf16* dst;
  switch (mat) {
    case 0: src = Wi;                  dst = Wi0T; break;
    case 1: src = Wh;                  dst = Wh0T; break;
    case 2: src = Wi + (size_t)H * H;  dst = Wi1T; break;
    default: src = Wh + (size_t)H * H; dst = Wh1T; break;
  }
  dst[(size_t)n * H + k] = (__bf16)src[(size_t)k * H + n];
}

// Broadcast initial hidden rows to bf16 [B][H] buffers; combined biases.
__global__ __launch_bounds__(256) void fill_init_kernel(
    const float* __restrict__ h0, const float* __restrict__ bi,
    const float* __restrict__ bh, __bf16* __restrict__ h0init,
    __bf16* __restrict__ h1init, float* __restrict__ c0,
    float* __restrict__ c1) {
  int i = blockIdx.x * blockDim.x + threadIdx.x;  // B*H
  int n = i & (H - 1);
  h0init[i] = (__bf16)h0[n];        // h0[0,0,:] broadcast over batch
  h1init[i] = (__bf16)h0[H + n];    // h0[0,1,:] broadcast over batch
  if (i < H) { c0[i] = bi[i] + bh[i]; c1[i] = bi[H + i] + bh[H + i]; }
}

// pre0 = x @ Wi0 + (bi0 + bh0), written in-place into d_out[b][t][:].
// x is [B,T,H] => flat [B*T][H] rows; output same flat layout.
// 16x32 output tile per wave, A converted f32->bf16 on the fly (once per 2 WMMA).
__global__ __launch_bounds__(256) void pregemm_kernel(
    const float* __restrict__ x, const __bf16* __restrict__ Wi0T,
    const float* __restrict__ c0, float* __restrict__ dout) {
  int wave = threadIdx.x >> 5, lane = threadIdx.x & 31;
  int r = lane & 15, half = lane >> 4;
  int tile = blockIdx.x * 8 + wave;        // (B*T/16) * (H/32) tiles
  int rt = tile >> 5, ct2 = tile & 31;
  int r0 = rt << 4, n0 = ct2 << 5;
  const float* xrow = x + (size_t)(r0 + r) * H;
  const __bf16* Bcol0 = Wi0T + (size_t)(n0 + r) * H;
  const __bf16* Bcol1 = Bcol0 + (size_t)16 * H;
  v8f acc0 = {0, 0, 0, 0, 0, 0, 0, 0};
  v8f acc1 = {0, 0, 0, 0, 0, 0, 0, 0};
#pragma unroll 2
  for (int k0 = 0; k0 < H; k0 += 32) {
    BFV a, b0, b1;
#pragma unroll
    for (int i = 0; i < 8; ++i) {          // on-the-fly f32 -> bf16 for A
      a.v[i]     = (__bf16)xrow[k0 + half * 16 + i];
      a.v[i + 8] = (__bf16)xrow[k0 + half * 16 + 8 + i];
    }
    b0.h[0] = *(const v8bf*)(Bcol0 + k0 + half * 16);
    b0.h[1] = *(const v8bf*)(Bcol0 + k0 + half * 16 + 8);
    b1.h[0] = *(const v8bf*)(Bcol1 + k0 + half * 16);
    b1.h[1] = *(const v8bf*)(Bcol1 + k0 + half * 16 + 8);
    acc0 = __builtin_amdgcn_wmma_f32_16x16x32_bf16(false, a.v, false, b0.v,
                                                   (short)0, acc0, false, false);
    acc1 = __builtin_amdgcn_wmma_f32_16x16x32_bf16(false, a.v, false, b1.v,
                                                   (short)0, acc1, false, false);
  }
#pragma unroll
  for (int j = 0; j < 2; ++j) {
    int col = n0 + j * 16 + r;
    float bias = c0[col];
    const v8f& acc = j ? acc1 : acc0;
#pragma unroll
    for (int v = 0; v < 8; ++v) {
      int row = r0 + half * 8 + v;
      dout[(size_t)row * H + col] = acc[v] + bias;
    }
  }
}

// Phase t: tiles [0,256)   compute h0_{t+1} = tanh(pre0[:,t+1] + h0_t @ Wh0)
//          tiles [256,512) compute h1_t = tanh(h0_t@Wi1 + h1_{t-1}@Wh1 + c1)
// Each tile is a 16x32 output slab handled by one wave (2 accumulators).
// Step 0 init: launch tiles [0,256) with t = -1 (pre column 0, A = h0init).
// Final step:  launch tiles [256,512) only (tile_base = 256).
__global__ __launch_bounds__(256) void rnn_step_kernel(
    float* __restrict__ dout,              // [B*T][H] (pre0 in, output out)
    float* __restrict__ hn,                // [B][L][H]
    const __bf16* __restrict__ h0A, const __bf16* __restrict__ h1A,
    __bf16* __restrict__ h0next, __bf16* __restrict__ h1cur,
    const __bf16* __restrict__ Wh0T, const __bf16* __restrict__ Wi1T,
    const __bf16* __restrict__ Wh1T, const float* __restrict__ c1,
    int t, int tile_base) {
  int wave = threadIdx.x >> 5, lane = threadIdx.x & 31;
  int r = lane & 15, half = lane >> 4;
  int tile = blockIdx.x * 8 + wave + tile_base;
  int st = tile & 255;
  int rt = st >> 5, ct2 = st & 31;
  int m0 = rt << 4, n0 = ct2 << 5;
  v8f acc0 = {0, 0, 0, 0, 0, 0, 0, 0};
  v8f acc1 = {0, 0, 0, 0, 0, 0, 0, 0};
  if (tile < 256) {  // ---- layer-0 next state
    wmma_bb2(h0A + (size_t)(m0 + r) * H, Wh0T + (size_t)(n0 + r) * H,
             &acc0, &acc1, half);
#pragma unroll
    for (int j = 0; j < 2; ++j) {
      int col = n0 + j * 16 + r;
      const v8f& acc = j ? acc1 : acc0;
#pragma unroll
      for (int v = 0; v < 8; ++v) {
        int row = m0 + half * 8 + v;  // batch index
        float pre = dout[((size_t)row * T + (size_t)(t + 1)) * H + col];
        float hv = tanhf(acc[v] + pre);
        h0next[(size_t)row * H + col] = (__bf16)hv;
        if (t + 1 == T - 1) hn[((size_t)row * L + 0) * H + col] = hv;
      }
    }
  } else {           // ---- layer-1 current state
    wmma_bb2(h0A + (size_t)(m0 + r) * H, Wi1T + (size_t)(n0 + r) * H,
             &acc0, &acc1, half);
    wmma_bb2(h1A + (size_t)(m0 + r) * H, Wh1T + (size_t)(n0 + r) * H,
             &acc0, &acc1, half);
#pragma unroll
    for (int j = 0; j < 2; ++j) {
      int col = n0 + j * 16 + r;
      float bias = c1[col];
      const v8f& acc = j ? acc1 : acc0;
#pragma unroll
      for (int v = 0; v < 8; ++v) {
        int row = m0 + half * 8 + v;
        float hv = tanhf(acc[v] + bias);
        dout[((size_t)row * T + (size_t)t) * H + col] = hv;
        h1cur[(size_t)row * H + col] = (__bf16)hv;
        if (t == T - 1) hn[((size_t)row * L + 1) * H + col] = hv;
      }
    }
  }
}

extern "C" void kernel_launch(void* const* d_in, const int* in_sizes, int n_in,
                              void* d_out, int out_size, void* d_ws, size_t ws_size,
                              hipStream_t stream) {
  (void)in_sizes; (void)n_in; (void)out_size; (void)ws_size;
  const float* x  = (const float*)d_in[0];   // [B,T,H]
  const float* h0 = (const float*)d_in[1];   // [1,L,H]
  const float* Wi = (const float*)d_in[2];   // [L,H,H]
  const float* bi = (const float*)d_in[3];   // [L,H]
  const float* Wh = (const float*)d_in[4];   // [L,H,H]
  const float* bh = (const float*)d_in[5];   // [L,H]

  float* dout = (float*)d_out;                      // [B*T*H] output
  float* hn   = dout + (size_t)B * T * H;           // [B*L*H] h_n

  // Workspace layout (~9.3 MB total)
  char* w = (char*)d_ws;
  size_t HHb = (size_t)H * H * sizeof(__bf16);
  __bf16* Wi0T = (__bf16*)w; w += HHb;
  __bf16* Wh0T = (__bf16*)w; w += HHb;
  __bf16* Wi1T = (__bf16*)w; w += HHb;
  __bf16* Wh1T = (__bf16*)w; w += HHb;
  float* c0 = (float*)w; w += H * sizeof(float);
  float* c1 = (float*)w; w += H * sizeof(float);
  size_t BHb = (size_t)B * H * sizeof(__bf16);
  __bf16* h0init  = (__bf16*)w; w += BHb;
  __bf16* h0buf0  = (__bf16*)w; w += BHb;
  __bf16* h0buf1  = (__bf16*)w; w += BHb;
  __bf16* h1buf0  = (__bf16*)w; w += BHb;
  __bf16* h1buf1  = (__bf16*)w; w += BHb;
  __bf16* h0buf[2] = {h0buf0, h0buf1};
  __bf16* h1buf[2] = {h1buf0, h1buf1};

  // One-time prep (parallel, off the critical path)
  convert_weights_kernel<<<(4u << 20) / 256, 256, 0, stream>>>(
      Wi, Wh, Wi0T, Wh0T, Wi1T, Wh1T);
  fill_init_kernel<<<(B * H) / 256, 256, 0, stream>>>(
      h0, bi, bh, h0init, h1buf[1], c0, c1);

  // Big parallel GEMM: pre0 -> stored in-place in d_out
  // (B*T/16) row tiles * (H/32) col slabs = 65536 wave-tiles, 8 waves/block
  pregemm_kernel<<<((B * T / 16) * (H / 32)) / 8, 256, 0, stream>>>(
      x, Wi0T, c0, dout);

  // Step 0 init: h0_0 = tanh(pre0[:,0] + h0init @ Wh0)  (G1 tiles only, t=-1)
  rnn_step_kernel<<<32, 256, 0, stream>>>(
      dout, hn, h0init, h0init, h0buf[0], h1buf[0],
      Wh0T, Wi1T, Wh1T, c1, -1, 0);

  // Recurrence: phase t produces h1_t (and h0_{t+1} while t < T-1)
  for (int t = 0; t < T; ++t) {
    const __bf16* h0A = h0buf[t & 1];
    const __bf16* h1A = h1buf[(t - 1) & 1];   // t==0 -> h1buf[1] == init
    __bf16* h0next = h0buf[(t + 1) & 1];
    __bf16* h1cur  = h1buf[t & 1];
    if (t < T - 1) {
      rnn_step_kernel<<<64, 256, 0, stream>>>(
          dout, hn, h0A, h1A, h0next, h1cur, Wh0T, Wi1T, Wh1T, c1, t, 0);
    } else {  // last step: layer-1 tiles only
      rnn_step_kernel<<<32, 256, 0, stream>>>(
          dout, hn, h0A, h1A, h0next, h1cur, Wh0T, Wi1T, Wh1T, c1, t, 256);
    }
  }
}
